// CachedMPS_68212670595935
// MI455X (gfx1250) — matrix operand order repaired and checked by reading, not compiled, not run on hardware
//
#include <hip/hip_runtime.h>

// MPS classifier: B=8192 batch, N=784 sites, D=64 bond, C=10 classes, F=2.
// Transposed contraction Mn^T = A^T x M^T on v_wmma_f32_16x16x32_f16.
// Site fragments are staged cooperatively into LDS (double-buffered) so each
// workgroup reads each site's 16KB from L2 exactly once; waves feed WMMA from
// ds_load_b128. Async-to-LDS (ASYNCcnt) path used when the toolchain exposes it.

typedef __attribute__((ext_vector_type(16))) _Float16 v16h;
typedef __attribute__((ext_vector_type(8)))  float    v8f;
typedef __attribute__((ext_vector_type(4)))  int      v4i;

#define BB 8192
#define NN 784
#define DD 64
#define CC 10
#define NMID 782
#define AWS_HALFS_PER_SITE 8192                    // 2(f) * 4(t) * 2(q) * 32(lane) * 16(e)
#define SITE_BYTES 16384
#define AWS_BYTES ((size_t)NMID * AWS_HALFS_PER_SITE * sizeof(_Float16))  // ~12.8 MB

#if defined(__has_builtin)
#if __has_builtin(__builtin_amdgcn_global_load_async_to_lds_b128) && \
    __has_builtin(__builtin_amdgcn_s_wait_asynccnt)
#define USE_ASYNC 1
#endif
#endif
#ifndef USE_ASYNC
#define USE_ASYNC 0
#endif

#if USE_ASYNC
#define AS1 __attribute__((address_space(1)))
#define AS3 __attribute__((address_space(3)))
#endif

// ---------------- prep 1: feature map, transposed to [n][b] float2(cos,sin) ----------------
__global__ void feat_kernel(const float* __restrict__ x, float2* __restrict__ fpk) {
  int idx = blockIdx.x * blockDim.x + threadIdx.x;
  if (idx >= NN * BB) return;
  int n = idx / BB;
  int b = idx - n * BB;
  float v = x[(size_t)b * NN + n];
  float s, c;
  sincosf(1.57079632679489662f * v, &s, &c);
  fpk[idx] = make_float2(c, s);
}

// ---------------- prep 2: swizzle cores_mid (fp32) into f16 WMMA A-fragment layout ----------
// layout: aws[s][f][t][q][lane][e], A-op row j = 16t + (lane&15),
//         K-local = e + 16*hi, global i = 32q + K-local,
//         value = cores_mid[s][f][i][j]   (i = contracted bond-in, j = bond-out)
__global__ void swizzle_kernel(const float* __restrict__ cm, _Float16* __restrict__ aws) {
  int idx = blockIdx.x * blockDim.x + threadIdx.x;
  if (idx >= NMID * AWS_HALFS_PER_SITE) return;
  int e    = idx & 15;
  int lane = (idx >> 4) & 31;
  int q    = (idx >> 9) & 1;
  int t    = (idx >> 10) & 3;
  int f    = (idx >> 12) & 1;
  int s    = idx >> 13;
  int r  = lane & 15;
  int hi = lane >> 4;
  int j  = t * 16 + r;
  int i  = 32 * q + 16 * hi + e;
  aws[idx] = (_Float16)cm[(((size_t)s * 2 + f) * 64 + i) * 64 + j];
}

// Build B operand (M^T, K=bond-in, N=batch col) from M in D-layout.
__device__ __forceinline__ void build_b(const float (&M)[4][8], int hi, v16h (&Bop)[2]) {
#pragma unroll
  for (int q = 0; q < 2; q++) {
    v16h b;
#pragma unroll
    for (int v = 0; v < 8; v++) {
      float nat = hi ? M[2 * q + 1][v] : M[2 * q][v];
      float oth = hi ? M[2 * q][v]     : M[2 * q + 1][v];
      float sh  = __shfl_xor(oth, 16, 32);
      float lo  = hi ? sh  : nat;
      float hh  = hi ? nat : sh;
      b[v]     = (_Float16)lo;
      b[v + 8] = (_Float16)hh;
    }
    Bop[q] = b;
  }
}

__global__ __launch_bounds__(256) void mps_main(
    const float*    __restrict__ core0,      // [2][64]
    const float*    __restrict__ core_last,  // [2][64][10]
    const float2*   __restrict__ fpk,        // [784][8192] (cos,sin)
    const _Float16* __restrict__ aws,        // [782][2][4][2][32][16] f16 A-fragments
    float*          __restrict__ out)        // [8192][10]
{
  __shared__ _Float16 lbuf[2][AWS_HALFS_PER_SITE];   // 2 x 16KB double buffer

  const int tid  = threadIdx.x;
  const int lane = tid & 31;
  const int wave = tid >> 5;
  const int c    = lane & 15;
  const int hi   = lane >> 4;
  const int bcol = blockIdx.x * 128 + wave * 16 + c;

  // ---- stage site 0 into buffer 0 (each thread moves 64B, lane-coalesced) ----
#if USE_ASYNC
  {
    AS1 char* g = (AS1 char*)(aws);
    AS3 char* l = (AS3 char*)(&lbuf[0][0]);
#pragma unroll
    for (int k = 0; k < 4; k++) {
      int off = (tid + 256 * k) * 16;
      __builtin_amdgcn_global_load_async_to_lds_b128((AS1 v4i*)(g + off),
                                                     (AS3 v4i*)(l + off), 0, 0);
    }
    __builtin_amdgcn_s_wait_asynccnt(0);
  }
#else
  {
    const uint4* g = (const uint4*)(aws);
    uint4* l = (uint4*)(&lbuf[0][0]);
#pragma unroll
    for (int k = 0; k < 4; k++) l[tid + 256 * k] = g[tid + 256 * k];
  }
#endif
  __syncthreads();

  // ---- site 0 init: M[j][b] = cos0*core0[0][j] + sin0*core0[1][j] (no norm, per ref) ----
  float M[4][8];
  {
    float2 f0 = fpk[bcol];
#pragma unroll
    for (int t = 0; t < 4; t++)
#pragma unroll
      for (int v = 0; v < 8; v++) {
        int j = t * 16 + hi * 8 + v;
        M[t][v] = f0.x * core0[j] + f0.y * core0[64 + j];
      }
  }

  v16h Bop[2];
  build_b(M, hi, Bop);

  // ---- sequential sweep over 782 middle sites ----
  for (int s = 0; s < NMID; s++) {
    const int p = s & 1;
    float2 fv = fpk[(size_t)(s + 1) * BB + bcol];

    // L2 prefetch two sites ahead
    __builtin_prefetch(aws + (size_t)(s + 2) * AWS_HALFS_PER_SITE + lane * 16, 0, 1);

#if USE_ASYNC
    // async-stage site s+1 into the other buffer (ASYNCcnt path, no VGPR round trip)
    if (s + 1 < NMID) {
      AS1 char* g = (AS1 char*)(aws + (size_t)(s + 1) * AWS_HALFS_PER_SITE);
      AS3 char* l = (AS3 char*)(&lbuf[p ^ 1][0]);
#pragma unroll
      for (int k = 0; k < 4; k++) {
        int off = (tid + 256 * k) * 16;
        __builtin_amdgcn_global_load_async_to_lds_b128((AS1 v4i*)(g + off),
                                                       (AS3 v4i*)(l + off), 0, 0);
      }
    }
#else
    // issue global loads for site s+1 now; ds_stores happen after compute
    uint4 st[4];
    if (s + 1 < NMID) {
      const uint4* g = (const uint4*)(aws + (size_t)(s + 1) * AWS_HALFS_PER_SITE);
#pragma unroll
      for (int k = 0; k < 4; k++) st[k] = g[tid + 256 * k];
    }
#endif

    // ---- compute site s from LDS buffer p ----
    const _Float16* lb = &lbuf[p][0] + lane * 16;
    float ssq = 0.0f;
#pragma unroll
    for (int t = 0; t < 4; t++) {
      v8f a0 = {};
      v8f a1 = {};
#pragma unroll
      for (int q = 0; q < 2; q++) {
        v16h af0 = *(const v16h*)(lb + (0 * 8 + t * 2 + q) * 512);
        a0 = __builtin_amdgcn_wmma_f32_16x16x32_f16(false, af0, false, Bop[q],
                                                    (short)0, a0, false, false);
      }
#pragma unroll
      for (int q = 0; q < 2; q++) {
        v16h af1 = *(const v16h*)(lb + (1 * 8 + t * 2 + q) * 512);
        a1 = __builtin_amdgcn_wmma_f32_16x16x32_f16(false, af1, false, Bop[q],
                                                    (short)0, a1, false, false);
      }
#pragma unroll
      for (int v = 0; v < 8; v++) {
        float m = fv.x * a0[v] + fv.y * a1[v];
        M[t][v] = m;
        ssq += m * m;
      }
    }
    // row L2 norm: batch column c lives in lanes c and c+16 only
    ssq += __shfl_xor(ssq, 16, 32);
    float inv = 1.0f / (sqrtf(ssq) + 1e-8f);
#pragma unroll
    for (int t = 0; t < 4; t++)
#pragma unroll
      for (int v = 0; v < 8; v++) M[t][v] *= inv;

    build_b(M, hi, Bop);

#if USE_ASYNC
    __builtin_amdgcn_s_wait_asynccnt(0);
#else
    if (s + 1 < NMID) {
      uint4* l = (uint4*)(&lbuf[p ^ 1][0]);
#pragma unroll
      for (int k = 0; k < 4; k++) l[tid + 256 * k] = st[k];
    }
#endif
    // publishes buffer p^1 AND protects buffer p before it is overwritten next iter
    __syncthreads();
  }

  // ---- last site: logits[b][cls] = sum_i M[i][b] * (cosL*CL[0][i][cls] + sinL*CL[1][i][cls]) ----
  float2 fl = fpk[(size_t)(NN - 1) * BB + bcol];
  float lg[CC];
#pragma unroll
  for (int cls = 0; cls < CC; cls++) lg[cls] = 0.0f;
#pragma unroll
  for (int t = 0; t < 4; t++)
#pragma unroll
    for (int v = 0; v < 8; v++) {
      int i = t * 16 + hi * 8 + v;
      float m = M[t][v];
#pragma unroll
      for (int cls = 0; cls < CC; cls++) {
        float w = fl.x * core_last[i * CC + cls] + fl.y * core_last[64 * CC + i * CC + cls];
        lg[cls] += m * w;
      }
    }
#pragma unroll
  for (int cls = 0; cls < CC; cls++) {
    float tot = lg[cls] + __shfl_xor(lg[cls], 16, 32);
    if (hi == 0) out[(size_t)bcol * CC + cls] = tot;
  }
}

extern "C" void kernel_launch(void* const* d_in, const int* in_sizes, int n_in,
                              void* d_out, int out_size, void* d_ws, size_t ws_size,
                              hipStream_t stream) {
  const float* x         = (const float*)d_in[0];   // [8192][784]
  const float* core0     = (const float*)d_in[1];   // [2][1][64]
  const float* cores_mid = (const float*)d_in[2];   // [782][2][64][64]
  const float* core_last = (const float*)d_in[3];   // [2][64][10]
  float* out = (float*)d_out;

  char* ws = (char*)d_ws;
  _Float16* aws = (_Float16*)ws;                    // 12.8 MB of f16 A-fragments
  float2*   fpk = (float2*)(ws + AWS_BYTES);        // 51.4 MB of (cos,sin) pairs, [n][b]

  {
    int total = NN * BB;
    feat_kernel<<<(total + 255) / 256, 256, 0, stream>>>(x, fpk);
  }
  {
    int total = NMID * AWS_HALFS_PER_SITE;
    swizzle_kernel<<<(total + 255) / 256, 256, 0, stream>>>(cores_mid, aws);
  }
  mps_main<<<BB / 128, 256, 0, stream>>>(core0, core_last, fpk, aws, out);
}
